// APRE_36180804501840
// MI455X (gfx1250) — matrix operand order repaired and checked by reading, not compiled, not run on hardware
//
#include <hip/hip_runtime.h>
#include <hip/hip_bf16.h>

// ---------------- problem constants ----------------
constexpr int NL    = 256;   // L (review length)
constexpr int NB    = 256;   // BERT dim
constexpr int NF    = 256;   // FD
constexpr int NA    = 20;    // aspects
constexpr int NC    = 100;   // conv channels
constexpr int NKH   = 3;     // conv kernel height
constexpr int NIM   = 868;   // 3*FD + C
constexpr int NREVS = 1024;
constexpr int NBATCH= 128;
constexpr int KCNN  = NKH * NF;      // 768
constexpr int KIMP  = 2 * NIM;       // 1736
constexpr int KIMPP = 1760;          // padded to mult of 32

// ---------------- WMMA types ----------------
typedef __attribute__((ext_vector_type(16))) __bf16 v16bf;
typedef __attribute__((ext_vector_type(8)))  __bf16 v8bf;
typedef __attribute__((ext_vector_type(8)))  float  v8f;

__device__ __forceinline__ v8f wmma_bf16(v16bf a, v16bf b, v8f c) {
  return __builtin_amdgcn_wmma_f32_16x16x32_bf16(false, a, false, b, (short)0, c, false, false);
}

__device__ __forceinline__ v16bf zero16() {
  v16bf z;
#pragma unroll
  for (int i = 0; i < 16; ++i) z[i] = (__bf16)0.f;
  return z;
}

__device__ __forceinline__ v8f zero8f() {
  v8f z;
#pragma unroll
  for (int i = 0; i < 8; ++i) z[i] = 0.f;
  return z;
}

// pack 16 consecutive f32 (as 4 float4) into a bf16 fragment
__device__ __forceinline__ v16bf pack16(float4 a, float4 b, float4 c, float4 d) {
  v16bf v;
  v[0]=(__bf16)a.x; v[1]=(__bf16)a.y; v[2]=(__bf16)a.z; v[3]=(__bf16)a.w;
  v[4]=(__bf16)b.x; v[5]=(__bf16)b.y; v[6]=(__bf16)b.z; v[7]=(__bf16)b.w;
  v[8]=(__bf16)c.x; v[9]=(__bf16)c.y; v[10]=(__bf16)c.z; v[11]=(__bf16)c.w;
  v[12]=(__bf16)d.x; v[13]=(__bf16)d.y; v[14]=(__bf16)d.z; v[15]=(__bf16)d.w;
  return v;
}

// A-fragment from global f32, row-major, K contiguous. rowp -> start of the
// 32-wide K chunk for this lane's matrix row. Layout (contiguous):
// K = half*16 + e  -> one contiguous run of 16.
__device__ __forceinline__ v16bf afrag_g(const float* __restrict__ rowp, int half) {
  const float4* p = (const float4*)rowp;
  return pack16(p[half*4], p[half*4+1], p[half*4+2], p[half*4+3]);
}

// fragment from two contiguous 8-element bf16 runs in LDS (16B aligned each)
__device__ __forceinline__ v16bf frag_2x8(const __bf16* plo, const __bf16* phi) {
  v8bf lo = *(const v8bf*)plo;
  v8bf hi = *(const v8bf*)phi;
  v16bf v;
#pragma unroll
  for (int i = 0; i < 8; ++i) { v[i] = lo[i]; v[8+i] = hi[i]; }
  return v;
}

// ---------------- K1: per-review fused kernel ----------------
// One block per (review, side). 256 threads = 8 waves.
// LDS: Wt[256][264] bf16 (W transposed, staged once) + UH[256][264] bf16.
constexpr int WT_STRIDE = 264;               // pad to kill bank conflicts
constexpr int UH_STRIDE = 264;
constexpr int WT_ELEMS  = 256 * WT_STRIDE;
constexpr int UH_ELEMS  = 256 * UH_STRIDE;
constexpr size_t K1_LDS_BYTES = (size_t)(WT_ELEMS + UH_ELEMS) * 2;

__global__ __launch_bounds__(256) void k1_per_review(
    const float* __restrict__ u_hid,  const float* __restrict__ i_hid,
    const float* __restrict__ u_pool, const float* __restrict__ i_pool,
    const float* __restrict__ u_loc,  const float* __restrict__ i_loc,
    const float* __restrict__ W_ub,   const float* __restrict__ b_ub,
    const float* __restrict__ W_ib,   const float* __restrict__ b_ib,
    const float* __restrict__ W_ucls, const float* __restrict__ W_icls,
    const float* __restrict__ K_u,    const float* __restrict__ b_ku,
    const float* __restrict__ K_i,    const float* __restrict__ b_ki,
    float* __restrict__ r_u, float* __restrict__ r_i,
    float* __restrict__ xim_u, float* __restrict__ xim_i)
{
  extern __shared__ char smem[];
  __bf16* Wt = (__bf16*)smem;                          // [f][k], stride WT_STRIDE
  __bf16* UH = (__bf16*)(smem + (size_t)WT_ELEMS * 2); // [l][f], stride UH_STRIDE

  const int n    = blockIdx.x;
  const int side = blockIdx.y;
  const float* Hn    = (side ? i_hid  : u_hid ) + (size_t)n * NL * NB;
  const float* Wmat  =  side ? W_ib   : W_ub;
  const float* bvec  =  side ? b_ib   : b_ub;
  const float* locn  = (side ? i_loc  : u_loc ) + (size_t)n * NA * NL;
  const float* Kc    =  side ? K_i    : K_u;     // [C][KH*FD] = [c][768]
  const float* bk    =  side ? b_ki   : b_ku;
  const float* pooln = (side ? i_pool : u_pool) + (size_t)n * NB;
  const float* Wcls  =  side ? W_icls : W_ucls;
  float* r_out = (side ? r_i : r_u) + (size_t)n * NA * NF;
  float* xim   = (side ? xim_i : xim_u) + (size_t)n * NIM;

  const int tid  = threadIdx.x;
  const int wave = tid >> 5;
  const int lane = tid & 31;
  const int mrow = lane & 15;     // matrix row (A) / col (B,C/D)
  const int half = lane >> 4;

  // ---- stage W^T as bf16 into LDS (coalesced f32 reads) ----
  for (int idx = tid; idx < NB * NF; idx += 256) {
    int k = idx >> 8, f = idx & 255;
    Wt[f * WT_STRIDE + k] = (__bf16)Wmat[idx];
  }
  __syncthreads();

  // ---- phase 1: UH = relu(H @ W + b), 256x256, kept in LDS as bf16 ----
  // wave owns 2 row-tiles x all 16 col-tiles; k-loop outer so A loads once.
  for (int mtt = 0; mtt < 2; ++mtt) {
    const int mt = wave * 2 + mtt;
    v8f acc[16];
#pragma unroll
    for (int i = 0; i < 16; ++i) acc[i] = zero8f();
    for (int kc = 0; kc < 8; ++kc) {
      v16bf a = afrag_g(Hn + (size_t)(mt * 16 + mrow) * NB + kc * 32, half);
#pragma unroll
      for (int nt = 0; nt < 16; ++nt) {
        const __bf16* bp = Wt + (nt * 16 + mrow) * WT_STRIDE + kc * 32 + half * 16;
        v16bf b = frag_2x8(bp, bp + 8);
        acc[nt] = wmma_bf16(a, b, acc[nt]);
      }
    }
#pragma unroll
    for (int nt = 0; nt < 16; ++nt) {
      const int f = nt * 16 + mrow;
      const float bias = bvec[f];
#pragma unroll
      for (int r = 0; r < 8; ++r) {
        const int l = mt * 16 + half * 8 + r;
        float v = acc[nt][r] + bias;
        v = v > 0.f ? v : 0.f;
        UH[l * UH_STRIDE + f] = (__bf16)v;
      }
    }
  }
  __syncthreads();   // UH ready for all consumers (Wt no longer needed)

  // ---- phase 2a: r = loc (20x256) @ UH (256x256); M padded 20->32 ----
  for (int t4 = 0; t4 < 4; ++t4) {
    const int tile = wave * 4 + t4;       // 32 tiles over 8 waves
    const int mt = tile >> 4, nt = tile & 15;
    v8f acc = zero8f();
    const int f = nt * 16 + mrow;
    for (int kc = 0; kc < 8; ++kc) {
      const int arow = mt * 16 + mrow;
      v16bf a = (arow < NA) ? afrag_g(locn + (size_t)arow * NL + kc * 32, half)
                            : zero16();
      const int kb = kc * 32 + half * 16;
      v16bf b;
#pragma unroll
      for (int e = 0; e < 16; ++e) b[e] = UH[(kb + e) * UH_STRIDE + f];
      acc = wmma_bf16(a, b, acc);
    }
#pragma unroll
    for (int r = 0; r < 8; ++r) {
      const int a_idx = mt * 16 + half * 8 + r;
      if (a_idx < NA) r_out[(size_t)a_idx * NF + f] = acc[r];
    }
  }

  // ---- phase 2b: CNN as GEMM: X(254x768) @ Kmat(768x100) -> max over t ----
  // X[t][kh*256+f] = UH[t+kh][f]. Waves 0..6 each own one 16-col strip.
  if (wave < 7) {
    const int nt = wave;
    const int c  = nt * 16 + mrow;
    v8f acc[16];
#pragma unroll
    for (int i = 0; i < 16; ++i) acc[i] = zero8f();
    for (int kc = 0; kc < 24; ++kc) {            // 768 / 32
      const int kh = kc >> 3;
      const int fb = (kc & 7) * 32;
      v16bf b;
      if (c < NC) {
        // Kmat[kk][c] = Kc[c*768 + kk]; kk = kc*32 + half*16 + e (contiguous)
        const float4* p4 = (const float4*)(Kc + (size_t)c * KCNN + kc * 32 + half * 16);
        b = pack16(p4[0], p4[1], p4[2], p4[3]);
      } else b = zero16();
#pragma unroll
      for (int mt = 0; mt < 16; ++mt) {
        const int t = mt * 16 + mrow;
        v16bf a;
        if (t < NL - NKH + 1) {
          const __bf16* up = UH + (t + kh) * UH_STRIDE + fb;
          a = frag_2x8(up + half * 16, up + half * 16 + 8);
        } else a = zero16();
        acc[mt] = wmma_bf16(a, b, acc[mt]);
      }
    }
    const float bias = (c < NC) ? bk[c] : 0.f;
    float mx = -1e30f;
#pragma unroll
    for (int mt = 0; mt < 16; ++mt) {
#pragma unroll
      for (int r = 0; r < 8; ++r) {
        const int t = mt * 16 + half * 8 + r;
        if (t < NL - NKH + 1) {
          float y = acc[mt][r] + bias;
          y = y > 0.f ? y : 0.f;
          mx = fmaxf(mx, y);
        }
      }
    }
    mx = fmaxf(mx, __shfl_xor(mx, 16, 32));   // combine the two half-lanes
    if (half == 0 && c < NC) xim[3 * NF + c] = mx;
  }

  // ---- phase 2c: avg / max / nonzero-count over L per feature ----
  {
    const int f = tid;
    float s = 0.f, mx = -1e30f;
    int cnt = 0;
    for (int l = 0; l < NL; ++l) {
      const float v = (float)UH[l * UH_STRIDE + f];
      s += v;
      mx = fmaxf(mx, v);
      cnt += (v != 0.f) ? 1 : 0;
    }
    xim[NF + f]     = s / (float)cnt;
    xim[2 * NF + f] = mx;
  }

  // ---- phase 2d: clsr = relu(pooler @ W_cls) ----
  {
    const int f = tid;
    float s = 0.f;
    for (int k = 0; k < NB; ++k) s += pooln[k] * Wcls[(size_t)k * NF + f];
    xim[f] = s > 0.f ? s : 0.f;
  }
}

// ---------------- K2: attention scores ----------------
__global__ void k2_scores_ex(const float* __restrict__ r_u, const float* __restrict__ r_i,
                             const float* __restrict__ u_emb, const float* __restrict__ i_emb,
                             const float* __restrict__ w_exu, const float* __restrict__ w_exi,
                             float* __restrict__ sc_ex /*[2][NREVS][NA]*/)
{
  const int side = blockIdx.y;
  const int idx  = blockIdx.x * blockDim.x + threadIdx.x;
  if (idx >= NREVS * NA) return;
  const int n = idx / NA, a = idx % NA;
  const float* r   = (side ? r_i : r_u) + (size_t)(n * NA + a) * NF;
  const float* emb = (side ? i_emb : u_emb) + (size_t)a * NF;
  const float* w   =  side ? w_exi : w_exu;
  float s2 = 0.f, dr = 0.f, de = 0.f;
  for (int f = 0; f < NF; ++f) {
    const float rv = r[f];
    s2 += rv;
    dr += rv * w[f];
    de += emb[f] * w[NF + f];
  }
  sc_ex[(size_t)(side * NREVS + n) * NA + a] = tanhf(s2 * (dr + de));
}

__global__ void k2_scores_im(const float* __restrict__ xim_u, const float* __restrict__ xim_i,
                             const float* __restrict__ w_imu, const float* __restrict__ w_imi,
                             float* __restrict__ sc_im /*[2][NREVS]*/)
{
  const int side = blockIdx.y;
  const int n = blockIdx.x * blockDim.x + threadIdx.x;
  if (n >= NREVS) return;
  const float* x = (side ? xim_i : xim_u) + (size_t)n * NIM;
  const float* w =  side ? w_imi : w_imu;
  float s = 0.f;
  for (int d = 0; d < NIM; ++d) s += x[d] * w[d];
  sc_im[side * NREVS + n] = tanhf(s);
}

// ---------------- K3: segment softmax + weighted sums -> concat buffers ----
__device__ __forceinline__ int lower_bound_i(const int* arr, int n, int key) {
  int lo = 0, hi = n;
  while (lo < hi) { int mid = (lo + hi) >> 1; if (arr[mid] < key) lo = mid + 1; else hi = mid; }
  return lo;
}

__global__ __launch_bounds__(256) void k3_segsum(
    const int* __restrict__ u_seg, const int* __restrict__ i_seg,
    const float* __restrict__ r_u, const float* __restrict__ r_i,
    const float* __restrict__ xim_u, const float* __restrict__ xim_i,
    const float* __restrict__ sc_ex, const float* __restrict__ sc_im,
    float* __restrict__ concat_ex /*[B*NA][512]*/,
    float* __restrict__ concat_im /*[B][KIMPP]*/)
{
  const int side = blockIdx.y, b = blockIdx.x, tid = threadIdx.x;
  const int* seg = side ? i_seg : u_seg;   // sorted
  const int n0 = lower_bound_i(seg, NREVS, b);
  const int n1 = lower_bound_i(seg, NREVS, b + 1);
  const float* rr = side ? r_i : r_u;
  const float* xx = side ? xim_i : xim_u;
  const float* se = sc_ex + (size_t)side * NREVS * NA;
  const float* si = sc_im + (size_t)side * NREVS;

  // explicit: per aspect, softmax weights over segment, weighted sum of r
  for (int a = 0; a < NA; ++a) {
    float m = -1e30f;
    for (int n = n0; n < n1; ++n) m = fmaxf(m, se[n * NA + a]);
    float z = 0.f;
    for (int n = n0; n < n1; ++n) z += expf(se[n * NA + a] - m);
    const int f = tid;
    float acc = 0.f;
    for (int n = n0; n < n1; ++n)
      acc += (expf(se[n * NA + a] - m) / z) * rr[(size_t)(n * NA + a) * NF + f];
    concat_ex[(size_t)(b * NA + a) * (2 * NF) + side * NF + f] = (n1 > n0) ? acc : 0.f;
  }

  // implicit
  float m = -1e30f;
  for (int n = n0; n < n1; ++n) m = fmaxf(m, si[n]);
  float z = 0.f;
  for (int n = n0; n < n1; ++n) z += expf(si[n] - m);
  for (int d = tid; d < NIM; d += blockDim.x) {
    float acc = 0.f;
    for (int n = n0; n < n1; ++n) acc += (expf(si[n] - m) / z) * xx[(size_t)n * NIM + d];
    concat_im[(size_t)b * KIMPP + side * NIM + d] = (n1 > n0) ? acc : 0.f;
  }
  if (side == 0)
    for (int d = KIMP + tid; d < KIMPP; d += blockDim.x)
      concat_im[(size_t)b * KIMPP + d] = 0.f;     // K padding
}

// ---------------- K4: generic WMMA GEMM  C = relu(A@B + bias), N = 256 ----
// A: [M][Kpad] f32 (ws, zero-padded cols); B: [Kreal][256] f32 (global, row guard)
__global__ __launch_bounds__(256) void k4_gemm_relu(
    const float* __restrict__ Amat, const float* __restrict__ Bmat,
    const float* __restrict__ bias, float* __restrict__ Cmat,
    int Kpad, int Kreal)
{
  const int tid = threadIdx.x;
  const int wave = tid >> 5, lane = tid & 31;
  const int nr = lane & 15, half = lane >> 4;
  const int row0 = blockIdx.x * 16;
  v8f acc[2];
  acc[0] = zero8f(); acc[1] = zero8f();
  const int nkc = Kpad / 32;
  for (int kc = 0; kc < nkc; ++kc) {
    v16bf a = afrag_g(Amat + (size_t)(row0 + nr) * Kpad + kc * 32, half);
#pragma unroll
    for (int j = 0; j < 2; ++j) {
      const int nt = wave * 2 + j;
      const int col = nt * 16 + nr;
      const int kb = kc * 32 + half * 16;
      v16bf b;
#pragma unroll
      for (int e = 0; e < 16; ++e) {
        const int kk = kb + e;
        b[e] = (__bf16)((kk < Kreal) ? Bmat[(size_t)kk * 256 + col] : 0.f);
      }
      acc[j] = wmma_bf16(a, b, acc[j]);
    }
  }
#pragma unroll
  for (int j = 0; j < 2; ++j) {
    const int col = wave * 32 + j * 16 + nr;
    const float bi = bias[col];
#pragma unroll
    for (int r = 0; r < 8; ++r) {
      const int row = row0 + half * 8 + r;
      float v = acc[j][r] + bi;
      Cmat[(size_t)row * 256 + col] = v > 0.f ? v : 0.f;
    }
  }
}

// ---------------- K5: final prediction ----------------
__global__ void k5_final(const int* __restrict__ uid, const int* __restrict__ iid,
                         const float* __restrict__ bu_tab, const float* __restrict__ bt_tab,
                         const float* __restrict__ ex_h, const float* __restrict__ im_h,
                         const float* __restrict__ W_ex2, const float* __restrict__ W_im2,
                         const float* __restrict__ gamma, float* __restrict__ out)
{
  const int b = threadIdx.x;
  if (b >= NBATCH) return;
  float p = bu_tab[uid[b]] + bt_tab[iid[b]];
  float s = 0.f;
  for (int f = 0; f < NF; ++f) s += im_h[(size_t)b * NF + f] * W_im2[f];
  p += s;
  for (int a = 0; a < NA; ++a) {
    float e = 0.f;
    for (int f = 0; f < NF; ++f) e += ex_h[(size_t)(b * NA + a) * NF + f] * W_ex2[f];
    p += e * gamma[a];
  }
  out[b] = p;
}

// ---------------- launch ----------------
extern "C" void kernel_launch(void* const* d_in, const int* in_sizes, int n_in,
                              void* d_out, int out_size, void* d_ws, size_t ws_size,
                              hipStream_t stream)
{
  (void)in_sizes; (void)n_in; (void)out_size; (void)ws_size;
  const float* u_out_hid = (const float*)d_in[0];
  const float* i_out_hid = (const float*)d_in[1];
  const float* u_pooler  = (const float*)d_in[2];
  const float* i_pooler  = (const float*)d_in[3];
  const float* urevs_loc = (const float*)d_in[4];
  const float* irevs_loc = (const float*)d_in[5];
  const int*   u_seg     = (const int*)d_in[6];
  const int*   i_seg     = (const int*)d_in[7];
  const int*   uid       = (const int*)d_in[8];
  const int*   iid       = (const int*)d_in[9];
  const float* u_emb_asp = (const float*)d_in[10];
  const float* i_emb_asp = (const float*)d_in[11];
  const float* W_ub      = (const float*)d_in[12];
  const float* b_ub      = (const float*)d_in[13];
  const float* W_ib      = (const float*)d_in[14];
  const float* b_ib      = (const float*)d_in[15];
  const float* w_exu     = (const float*)d_in[16];
  const float* w_exi     = (const float*)d_in[17];
  const float* w_imu     = (const float*)d_in[18];
  const float* w_imi     = (const float*)d_in[19];
  const float* W_ucls    = (const float*)d_in[20];
  const float* W_icls    = (const float*)d_in[21];
  const float* K_u       = (const float*)d_in[22];
  const float* b_ku      = (const float*)d_in[23];
  const float* K_i       = (const float*)d_in[24];
  const float* b_ki      = (const float*)d_in[25];
  const float* W_ex1     = (const float*)d_in[26];
  const float* b_ex1     = (const float*)d_in[27];
  const float* W_ex2     = (const float*)d_in[28];
  const float* W_im1     = (const float*)d_in[29];
  const float* b_im1     = (const float*)d_in[30];
  const float* W_im2     = (const float*)d_in[31];
  const float* bu_tab    = (const float*)d_in[32];
  const float* bt_tab    = (const float*)d_in[33];
  const float* gamma     = (const float*)d_in[34];

  char* ws = (char*)d_ws;
  size_t off = 0;
  auto carve = [&](size_t bytes) -> char* {
    char* p = ws + off;
    off = (off + bytes + 255) & ~(size_t)255;
    return p;
  };
  float* r_u       = (float*)carve((size_t)NREVS * NA * NF * 4);
  float* r_i       = (float*)carve((size_t)NREVS * NA * NF * 4);
  float* xim_u     = (float*)carve((size_t)NREVS * NIM * 4);
  float* xim_i     = (float*)carve((size_t)NREVS * NIM * 4);
  float* sc_ex     = (float*)carve((size_t)2 * NREVS * NA * 4);
  float* sc_im     = (float*)carve((size_t)2 * NREVS * 4);
  float* concat_ex = (float*)carve((size_t)NBATCH * NA * 2 * NF * 4);
  float* concat_im = (float*)carve((size_t)NBATCH * KIMPP * 4);
  float* ex_h      = (float*)carve((size_t)NBATCH * NA * NF * 4);
  float* im_h      = (float*)carve((size_t)NBATCH * NF * 4);

  hipFuncSetAttribute((const void*)k1_per_review,
                      hipFuncAttributeMaxDynamicSharedMemorySize,
                      (int)K1_LDS_BYTES);

  k1_per_review<<<dim3(NREVS, 2), 256, K1_LDS_BYTES, stream>>>(
      u_out_hid, i_out_hid, u_pooler, i_pooler, urevs_loc, irevs_loc,
      W_ub, b_ub, W_ib, b_ib, W_ucls, W_icls, K_u, b_ku, K_i, b_ki,
      r_u, r_i, xim_u, xim_i);

  k2_scores_ex<<<dim3((NREVS * NA + 255) / 256, 2), 256, 0, stream>>>(
      r_u, r_i, u_emb_asp, i_emb_asp, w_exu, w_exi, sc_ex);
  k2_scores_im<<<dim3((NREVS + 255) / 256, 2), 256, 0, stream>>>(
      xim_u, xim_i, w_imu, w_imi, sc_im);

  k3_segsum<<<dim3(NBATCH, 2), 256, 0, stream>>>(
      u_seg, i_seg, r_u, r_i, xim_u, xim_i, sc_ex, sc_im, concat_ex, concat_im);

  // ex_h = relu(concat_ex(2560x512) @ W_ex1(512x256) + b_ex1)
  k4_gemm_relu<<<dim3(NBATCH * NA / 16), 256, 0, stream>>>(
      concat_ex, W_ex1, b_ex1, ex_h, 2 * NF, 2 * NF);
  // im_h = relu(concat_im(128x1760) @ W_im1(1736x256) + b_im1)
  k4_gemm_relu<<<dim3(NBATCH / 16), 256, 0, stream>>>(
      concat_im, W_im1, b_im1, im_h, KIMPP, KIMP);

  k5_final<<<dim3(1), 128, 0, stream>>>(
      uid, iid, bu_tab, bt_tab, ex_h, im_h, W_ex2, W_im2, gamma, (float*)d_out);
}
